// multiHead_dec_44006234915477
// MI455X (gfx1250) — matrix operand order repaired and checked by reading, not compiled, not run on hardware
//
#include <hip/hip_runtime.h>
#include <hip/hip_bf16.h>
#include <math.h>

typedef __bf16 bf16_t;
typedef __attribute__((ext_vector_type(2)))  __bf16 v2bf;
typedef __attribute__((ext_vector_type(4)))  __bf16 v4bf;
typedef __attribute__((ext_vector_type(8)))  __bf16 v8bf;
typedef __attribute__((ext_vector_type(16))) __bf16 v16bf;
typedef __attribute__((ext_vector_type(8)))  float  v8f;
typedef __attribute__((ext_vector_type(4)))  int    v4i;

#define AS1 __attribute__((address_space(1)))
#define AS3 __attribute__((address_space(3)))

#if defined(__has_builtin)
#if __has_builtin(__builtin_amdgcn_global_load_async_to_lds_b128) && \
    __has_builtin(__builtin_amdgcn_s_wait_asynccnt)
#define USE_ASYNC_LDS 1
#endif
#endif

#define WMMA_BF16(a, b, c) \
  __builtin_amdgcn_wmma_f32_16x16x32_bf16(false, (a), false, (b), (short)0, (c), false, false)

static __device__ __forceinline__ v16bf frag16(const bf16_t* p0, const bf16_t* p1) {
  v8bf lo = *(const v8bf*)p0;
  v8bf hi = *(const v8bf*)p1;
  return __builtin_shufflevector(lo, hi, 0,1,2,3,4,5,6,7,8,9,10,11,12,13,14,15);
}

static __device__ __forceinline__ v8f v8f_zero() {
  v8f z = {0.f,0.f,0.f,0.f,0.f,0.f,0.f,0.f};
  return z;
}

static __device__ __forceinline__ float fast_exp2(float x) {
#if defined(__has_builtin)
#if __has_builtin(__builtin_amdgcn_exp2f)
  return __builtin_amdgcn_exp2f(x);
#else
  return exp2f(x);
#endif
#else
  return exp2f(x);
#endif
}

// ---------------------------------------------------------------------------
// Projection GEMM: Y[m,n] = sum_k X[m,k] * W[k,n], fp32 in, bf16 out.
// BM=128, BN=128, BK=32, 256 threads = 8 waves (2 M-groups x 4 N-groups).
// ---------------------------------------------------------------------------
__launch_bounds__(256)
__global__ void proj_gemm_bf16(const float* __restrict__ X, const float* __restrict__ W,
                               bf16_t* __restrict__ Y, int M, int Kdim, int N,
                               int transpose_out, int Trows) {
  constexpr int LDT = 40;  // padded LDS stride (80 B, conflict-free b128)
  __shared__ bf16_t As[128 * LDT];  // [m][k]
  __shared__ bf16_t Bs[128 * LDT];  // [n][k]  (W tile transposed)

  const int t  = threadIdx.x;
  const int m0 = blockIdx.y * 128;
  const int n0 = blockIdx.x * 128;

  const int lane = t & 31;
  const int wv   = t >> 5;
  const int lo   = lane & 15;
  const bool hi  = lane >= 16;
  const int wm   = (wv & 1) * 64;
  const int wn   = (wv >> 1) * 32;

  // staging assignments
  const int ar  = t >> 3;         // A: rows {ar, ar+32, ar+64, ar+96}
  const int ac  = (t & 7) * 4;    // A: 4 cols
  const int bk2 = (t >> 4) * 2;   // B: k-pair 0,2,...,30
  const int bn8 = (t & 15) * 8;   // B: 8 cols of n

  v8f acc[4][2];
#pragma unroll
  for (int mi = 0; mi < 4; ++mi)
#pragma unroll
    for (int ni = 0; ni < 2; ++ni) acc[mi][ni] = v8f_zero();

  for (int k0 = 0; k0 < Kdim; k0 += 32) {
    // stage A (fp32 -> bf16)
#pragma unroll
    for (int i = 0; i < 4; ++i) {
      int r = ar + 32 * i;
      float4 f = *(const float4*)(X + (size_t)(m0 + r) * Kdim + k0 + ac);
      v4bf h = { (bf16_t)f.x, (bf16_t)f.y, (bf16_t)f.z, (bf16_t)f.w };
      *(v4bf*)(As + r * LDT + ac) = h;
    }
    // stage B transposed, packed k-pairs (8x b32 stores instead of 16x b16)
    {
      const float* w0 = W + (size_t)(k0 + bk2) * N + n0 + bn8;
      float4 a0 = *(const float4*)(w0);
      float4 a1 = *(const float4*)(w0 + 4);
      float4 b0 = *(const float4*)(w0 + N);
      float4 b1 = *(const float4*)(w0 + N + 4);
      float klo[8] = {a0.x, a0.y, a0.z, a0.w, a1.x, a1.y, a1.z, a1.w};
      float khi[8] = {b0.x, b0.y, b0.z, b0.w, b1.x, b1.y, b1.z, b1.w};
#pragma unroll
      for (int j = 0; j < 8; ++j) {
        v2bf h = { (bf16_t)klo[j], (bf16_t)khi[j] };
        *(v2bf*)(Bs + (bn8 + j) * LDT + bk2) = h;
      }
    }
    __syncthreads();

    v16bf afr[4], bfr[2];
#pragma unroll
    for (int mi = 0; mi < 4; ++mi) {
      int row = wm + mi * 16 + lo;
      int kA0 = hi ? 16 : 0, kA1 = hi ? 24 : 8;
      afr[mi] = frag16(As + row * LDT + kA0, As + row * LDT + kA1);
    }
#pragma unroll
    for (int ni = 0; ni < 2; ++ni) {
      int col = wn + ni * 16 + lo;
      int kB0 = hi ? 16 : 0;
      bfr[ni] = frag16(Bs + col * LDT + kB0, Bs + col * LDT + kB0 + 8);
    }
#pragma unroll
    for (int mi = 0; mi < 4; ++mi)
#pragma unroll
      for (int ni = 0; ni < 2; ++ni)
        acc[mi][ni] = WMMA_BF16(afr[mi], bfr[ni], acc[mi][ni]);
    __syncthreads();
  }

  int bb = 0, sbase = 0;
  if (transpose_out) { bb = m0 / Trows; sbase = m0 - bb * Trows; }
#pragma unroll
  for (int mi = 0; mi < 4; ++mi)
#pragma unroll
    for (int ni = 0; ni < 2; ++ni)
#pragma unroll
      for (int vg = 0; vg < 8; ++vg) {
        int row = wm + mi * 16 + vg + (hi ? 8 : 0);
        int gn  = n0 + wn + ni * 16 + lo;
        bf16_t v = (bf16_t)acc[mi][ni][vg];
        if (!transpose_out)
          Y[(size_t)(m0 + row) * N + gn] = v;
        else
          Y[((size_t)bb * N + gn) * (size_t)Trows + (sbase + row)] = v;
      }
}

// ---------------------------------------------------------------------------
// lengths[b] = count of nonzero energy[b,0,:]
// ---------------------------------------------------------------------------
__global__ void init_len(int* len) {
  if (threadIdx.x < 8) len[threadIdx.x] = 0;
}

__global__ void len_kernel(const bf16_t* __restrict__ Qb, const bf16_t* __restrict__ Kb,
                           int* __restrict__ len, int Tq, int Tk, int Dd) {
  int b = blockIdx.y;
  int k = blockIdx.x * blockDim.x + threadIdx.x;
  const bf16_t* q  = Qb + (size_t)b * Tq * Dd;  // query row 0
  const bf16_t* kr = Kb + ((size_t)b * Tk + k) * Dd;
  float s = 0.f;
  for (int d = 0; d < Dd; ++d) s += (float)q[d] * (float)kr[d];
  if (s != 0.0f) atomicAdd(len + b, 1);
}

// ---------------------------------------------------------------------------
// Flash attention: one workgroup (8 waves) per (batch, 16-row q-tile).
// ---------------------------------------------------------------------------
__launch_bounds__(256)
__global__ void flash_attn(const bf16_t* __restrict__ Qb, const bf16_t* __restrict__ Kb,
                           const bf16_t* __restrict__ Vt, const int* __restrict__ len,
                           float* __restrict__ Out) {
  constexpr int Tq = 2048, Tk = 2048, Dd = 1024, BKV = 128;
  constexpr int QLD = 1032;  // padded strides (conflict-free b128/b128)
  constexpr int SLD = 132;   // float stride: 16B-aligned float4 rows, bank-spread
  constexpr int PLD = 136;
  __shared__ bf16_t Qs[16 * QLD];
  __shared__ float  Ssh[16 * SLD];
  __shared__ bf16_t Psh[16 * PLD];

  const int t    = threadIdx.x;
  const int lane = t & 31;
  const int wv   = t >> 5;
  const int lo   = lane & 15;
  const bool hi  = lane >= 16;
  const int b    = blockIdx.y;
  const int q0   = blockIdx.x * 16;
  const int L    = len[b];
  // softmax computed in exp2 domain: s2 = S * (1/sqrt(64) * log2(e))
  const float sl2 = 0.125f * 1.44269504088896340736f;

  // ---- stage Q tile (16 x 1024 bf16) ----
  {
    int row = t >> 4;
    int col = (t & 15) * 64;
    const bf16_t* src = Qb + (size_t)(b * Tq + q0 + row) * Dd + col;
    bf16_t* dst = Qs + row * QLD + col;
#ifdef USE_ASYNC_LDS
#pragma unroll
    for (int j = 0; j < 8; ++j)
      __builtin_amdgcn_global_load_async_to_lds_b128(
          (AS1 v4i*)(src + j * 8), (AS3 v4i*)(dst + j * 8), 0, 0);
    __builtin_amdgcn_s_wait_asynccnt(0);
#else
#pragma unroll
    for (int j = 0; j < 8; ++j)
      *(uint4*)(dst + j * 8) = *(const uint4*)(src + j * 8);
#endif
  }
  __syncthreads();

  v8f Oacc[8];
#pragma unroll
  for (int nt = 0; nt < 8; ++nt) Oacc[nt] = v8f_zero();
  float m_run = -__builtin_inff();  // exp2-domain running max (row = lane&15)
  float l_run = 0.f;

  const bf16_t* Kbase = Kb + (size_t)b * Tk * Dd;
  const bf16_t* Vbase = Vt + (size_t)b * Dd * Tk;

  for (int kv0 = 0; kv0 < Tk; kv0 += BKV) {
    // ---- S = Q · K^T : wave wv computes columns [kv0+16*wv, +16) ----
    const int kvc = kv0 + wv * 16;
    const bf16_t* krow = Kbase + (size_t)(kvc + lo) * Dd;
    // prefetch next block's K row and V rows for this wave into near caches
    if (kv0 + BKV < Tk) {
      __builtin_prefetch(krow + (size_t)BKV * Dd, 0, 1);
      __builtin_prefetch(Vbase + (size_t)(wv * 128 + lo) * Tk + kv0 + BKV, 0, 1);
    }
    v8f Sacc = v8f_zero();
#pragma unroll 4
    for (int d0 = 0; d0 < Dd; d0 += 32) {
      v16bf a = frag16(Qs + lo * QLD + d0 + (hi ? 16 : 0),
                       Qs + lo * QLD + d0 + (hi ? 24 : 8));
      const bf16_t* kp = krow + d0 + (hi ? 16 : 0);
      v16bf bf = frag16(kp, kp + 8);
      Sacc = WMMA_BF16(a, bf, Sacc);
    }
#pragma unroll
    for (int vg = 0; vg < 8; ++vg) {
      int row = vg + (hi ? 8 : 0);
      Ssh[row * SLD + wv * 16 + lo] = Sacc[vg];
    }
    __syncthreads();

    // ---- online softmax stats (lane handles row = lane&15), float4 reads ----
    float mb = -__builtin_inff();
#pragma unroll 4
    for (int c = 0; c < BKV; c += 4) {
      float4 s4 = *(const float4*)(Ssh + lo * SLD + c);
      float v0 = ((kv0 + c + 0) < L) ? s4.x * sl2 : -__builtin_inff();
      float v1 = ((kv0 + c + 1) < L) ? s4.y * sl2 : -__builtin_inff();
      float v2 = ((kv0 + c + 2) < L) ? s4.z * sl2 : -__builtin_inff();
      float v3 = ((kv0 + c + 3) < L) ? s4.w * sl2 : -__builtin_inff();
      mb = fmaxf(mb, fmaxf(fmaxf(v0, v1), fmaxf(v2, v3)));
    }
    float m_new = fmaxf(m_run, mb);
    float psum = 0.f;
#pragma unroll 4
    for (int c = 0; c < BKV; c += 4) {
      float4 s4 = *(const float4*)(Ssh + lo * SLD + c);
      if ((kv0 + c + 0) < L) psum += fast_exp2(s4.x * sl2 - m_new);
      if ((kv0 + c + 1) < L) psum += fast_exp2(s4.y * sl2 - m_new);
      if ((kv0 + c + 2) < L) psum += fast_exp2(s4.z * sl2 - m_new);
      if ((kv0 + c + 3) < L) psum += fast_exp2(s4.w * sl2 - m_new);
    }
    float alpha = (m_run > -__builtin_inff()) ? fast_exp2(m_run - m_new) : 0.f;
    l_run = l_run * alpha + psum;
    m_run = m_new;

    float alphav[8], mnewv[8];
#pragma unroll
    for (int vg = 0; vg < 8; ++vg) {
      int row = vg + (hi ? 8 : 0);
      alphav[vg] = __shfl(alpha, row, 32);
      mnewv[vg]  = __shfl(m_new, row, 32);
    }
#pragma unroll
    for (int nt = 0; nt < 8; ++nt)
#pragma unroll
      for (int vg = 0; vg < 8; ++vg) Oacc[nt][vg] *= alphav[vg];

    // P for this wave's own S tile -> LDS (bf16)
    {
      bool valid = (kvc + lo) < L;
#pragma unroll
      for (int vg = 0; vg < 8; ++vg) {
        int row = vg + (hi ? 8 : 0);
        float p = valid ? fast_exp2(Sacc[vg] * sl2 - mnewv[vg]) : 0.f;
        Psh[row * PLD + wv * 16 + lo] = (bf16_t)p;
      }
    }
    __syncthreads();

    // ---- O += P · V : wave wv owns d in [wv*128, wv*128+128) ----
#pragma unroll
    for (int kk = 0; kk < 4; ++kk) {
      int kc = kk * 32;
      v16bf a = frag16(Psh + lo * PLD + kc + (hi ? 16 : 0),
                       Psh + lo * PLD + kc + (hi ? 24 : 8));
#pragma unroll
      for (int nt = 0; nt < 8; ++nt) {
        int d = wv * 128 + nt * 16 + lo;
        const bf16_t* vp = Vbase + (size_t)d * Tk + kv0 + kc + (hi ? 16 : 0);
        v16bf bf = frag16(vp, vp + 8);
        Oacc[nt] = WMMA_BF16(a, bf, Oacc[nt]);
      }
    }
    __syncthreads();
  }

  // ---- epilogue: O / l, store fp32 ----
  float linv[8];
#pragma unroll
  for (int vg = 0; vg < 8; ++vg) {
    int row = vg + (hi ? 8 : 0);
    float lr = __shfl(l_run, row, 32);
    linv[vg] = (lr > 0.f) ? 1.f / lr : 0.f;
  }
#pragma unroll
  for (int nt = 0; nt < 8; ++nt) {
    int d = wv * 128 + nt * 16 + lo;
#pragma unroll
    for (int vg = 0; vg < 8; ++vg) {
      int row = vg + (hi ? 8 : 0);
      Out[(size_t)(b * Tq + q0 + row) * Dd + d] = Oacc[nt][vg] * linv[vg];
    }
  }
}

// ---------------------------------------------------------------------------
extern "C" void kernel_launch(void* const* d_in, const int* in_sizes, int n_in,
                              void* d_out, int out_size, void* d_ws, size_t ws_size,
                              hipStream_t stream) {
  const float* x   = (const float*)d_in[0];
  const float* ctx = (const float*)d_in[1];
  const float* Wq  = (const float*)d_in[2];
  const float* Wk  = (const float*)d_in[3];
  const float* Wv  = (const float*)d_in[4];
  float* out = (float*)d_out;

  const int Bb = 8, Tq = 2048, Tk = 2048, E = 1024, Dd = 1024;

  char* ws = (char*)d_ws;
  bf16_t* Qb = (bf16_t*)ws;                           // [B][Tq][D]  32 MB
  bf16_t* Kb = (bf16_t*)(ws + ((size_t)32 << 20));    // [B][Tk][D]  32 MB
  bf16_t* Vt = (bf16_t*)(ws + ((size_t)64 << 20));    // [B][D][Tk]  32 MB
  int*    ln = (int*)   (ws + ((size_t)96 << 20));    // [B]

  dim3 blk(256);
  dim3 gproj(Dd / 128, (Bb * Tq) / 128);  // (8, 128)

  proj_gemm_bf16<<<gproj, blk, 0, stream>>>(x,   Wq, Qb, Bb * Tq, E, Dd, 0, Tk);
  proj_gemm_bf16<<<gproj, blk, 0, stream>>>(ctx, Wk, Kb, Bb * Tk, E, Dd, 0, Tk);
  proj_gemm_bf16<<<gproj, blk, 0, stream>>>(ctx, Wv, Vt, Bb * Tk, E, Dd, 1, Tk);

  init_len<<<dim3(1), dim3(32), 0, stream>>>(ln);
  len_kernel<<<dim3(Tk / 256, Bb), blk, 0, stream>>>(Qb, Kb, ln, Tq, Tk, Dd);

  flash_attn<<<dim3(Tq / 16, Bb), blk, 0, stream>>>(Qb, Kb, Vt, ln, out);
}
